// SNNEMotionNet_24352464569648
// MI455X (gfx1250) — matrix operand order repaired and checked by reading, not compiled, not run on hardware
//
#include <hip/hip_runtime.h>
#include <hip/hip_bf16.h>

// ---------------------------------------------------------------------------
// SNN pipeline for MI455X (gfx1250, wave32, WMMA).
//   conv1+bn1  -> GEMM  [32256 x 2048] x [2048 x 256]  (bf16 WMMA, f32 acc)
//   conv2+bn2  -> GEMM  [32256 x  768] x [ 768 x 256]
//   LIF scan   -> spikes (bf16 0/1)
//   dense w1   -> GEMM  [32256 x  256] x [ 256 x 256]
//   LIF scan   -> spikes
//   dense w2   -> wave-reduction + LIF scan -> d_out [512,2,63] f32
//
// GEMM blocks: 256 threads = 8 waves, 32x128 output tile, each wave owns two
// 16x16 accumulators (M and M+16) sharing B fragments. A panels (32 rows x
// 128 K, bf16) staged in padded LDS once per 4 K-subchunks -> 8 WMMAs per
// barrier pair.
// ---------------------------------------------------------------------------

#define BATCH 512
#define CIN   64
#define COUT  256
#define TSTEP 63
#define ROWS  (BATCH * TSTEP)   // 32256
#define K1TOT 2048               // CIN * 32
#define K2TOT 768                // 3 * COUT
#define LDSP  136                // 128 + 8 pad (272B row stride -> conflict-free)

typedef __attribute__((ext_vector_type(16))) __bf16          v16bf;
typedef __attribute__((ext_vector_type(8)))  float           v8f;
typedef __attribute__((ext_vector_type(8)))  unsigned short  v8u16;

struct HalfPair { v8u16 lo, hi; };

__device__ __forceinline__ unsigned short f2bf(float f) {
    unsigned int u = __float_as_uint(f);
    u += 0x7FFFu + ((u >> 16) & 1u);           // round-to-nearest-even
    return (unsigned short)(u >> 16);
}

__device__ __forceinline__ v16bf packbf(v8u16 lo, v8u16 hi) {
    HalfPair p{lo, hi};
    return __builtin_bit_cast(v16bf, p);
}

__device__ __forceinline__ v8u16 cvt8(v8f fv) {
    v8u16 r;
#pragma unroll
    for (int j = 0; j < 8; ++j) r[j] = f2bf(fv[j]);
    return r;
}

// ---------------------------------------------------------------------------
// Weight prep: fold BN into conv weights, convert to bf16 B-matrix layout
// WB[o][k] contiguous in k (so a B fragment = 2x16B contiguous loads).
// ---------------------------------------------------------------------------
__global__ void prep_conv1(const float* __restrict__ w, const float* __restrict__ b,
                           const float* __restrict__ g, const float* __restrict__ beta,
                           const float* __restrict__ m, const float* __restrict__ v,
                           unsigned short* __restrict__ WB, float* __restrict__ bias) {
    int tid = blockIdx.x * 256 + threadIdx.x;                 // 256*2048
    int o = tid >> 11, kc = tid & 2047;
    int ci = kc >> 5, k = kc & 31;
    float scale = g[o] * __frsqrt_rn(v[o] + 1e-5f);
    WB[tid] = f2bf(w[((size_t)o * CIN + ci) * 32 + k] * scale);
    if (kc == 0) bias[o] = (b[o] - m[o]) * scale + beta[o];
}

__global__ void prep_conv2(const float* __restrict__ w, const float* __restrict__ b,
                           const float* __restrict__ g, const float* __restrict__ beta,
                           const float* __restrict__ m, const float* __restrict__ v,
                           unsigned short* __restrict__ WB, float* __restrict__ bias) {
    int tid = blockIdx.x * 256 + threadIdx.x;                 // 256*768
    int o = tid / K2TOT, kc = tid - o * K2TOT;
    int dk = kc >> 8, ci = kc & 255;
    float scale = g[o] * __frsqrt_rn(v[o] + 1e-5f);
    WB[tid] = f2bf(w[((size_t)o * COUT + ci) * 3 + dk] * scale);
    if (kc == 0) bias[o] = (b[o] - m[o]) * scale + beta[o];
}

__global__ void prep_dense(const float* __restrict__ w1, unsigned short* __restrict__ WB) {
    int tid = blockIdx.x * 256 + threadIdx.x;                 // 256*256
    WB[tid] = f2bf(w1[tid]);
}

// ---------------------------------------------------------------------------
// GEMM1: conv1 as implicit GEMM. A row (n,t), A col (ci*32+k) = x[n][ci][16t+k].
// 16 K-panels of 128; 8 WMMAs per barrier pair.
// ---------------------------------------------------------------------------
__global__ void gemm1_kernel(const float* __restrict__ x,
                             const unsigned short* __restrict__ WB,
                             const float* __restrict__ bias,
                             float* __restrict__ Z) {
    __shared__ __align__(16) unsigned short As[32][LDSP];
    const int tid  = threadIdx.x;
    const int lane = tid & 31, wave = tid >> 5;
    const int rowBase = blockIdx.x * 32;
    const int coBase  = blockIdx.y * 128 + wave * 16;
    const int m = lane & 15, h = lane >> 4;
    const int col = coBase + m;                         // B column / D column

    // cooperative A loader coordinates (f32 -> bf16 into LDS, 16 elems/thread)
    const int rl = tid >> 3;                            // 0..31 panel row
    const int kq = (tid & 7) * 16;                      // 0..112
    const int growA = rowBase + rl;
    const int nA = growA / TSTEP, tA = growA - nA * TSTEP;
    const float* xrow = x + ((size_t)nA * CIN) * 1024 + tA * 16;

    v8f acc0 = {}, acc1 = {};
    for (int g2 = 0; g2 < 16; ++g2) {                   // 128-wide K panels
        __syncthreads();
#pragma unroll
        for (int half = 0; half < 2; ++half) {
            const int kglob = g2 * 128 + kq + half * 8;
            const int ci = kglob >> 5, k = kglob & 31;
            v8f fv = *(const v8f*)(xrow + (size_t)ci * 1024 + k);
            *(v8u16*)&As[rl][kq + half * 8] = cvt8(fv);
        }
        if (g2 < 15) {
            const int kn = (g2 + 1) * 128 + kq;
            __builtin_prefetch(xrow + (size_t)(kn >> 5) * 1024 + (kn & 31), 0, 1);
        }
        __syncthreads();
#pragma unroll
        for (int s = 0; s < 4; ++s) {
            const int kk = s * 32;
            v8u16 a0 = *(const v8u16*)&As[m][kk + 8 * h];
            v8u16 a1 = *(const v8u16*)&As[m][kk + 16 + 8 * h];
            v8u16 c0 = *(const v8u16*)&As[16 + m][kk + 8 * h];
            v8u16 c1 = *(const v8u16*)&As[16 + m][kk + 16 + 8 * h];
            const unsigned short* bp = WB + (size_t)col * K1TOT + g2 * 128 + kk + 8 * h;
            v16bf bv = packbf(*(const v8u16*)bp, *(const v8u16*)(bp + 16));
            acc0 = __builtin_amdgcn_wmma_f32_16x16x32_bf16(
                false, packbf(a0, a1), false, bv, (short)0, acc0, false, false);
            acc1 = __builtin_amdgcn_wmma_f32_16x16x32_bf16(
                false, packbf(c0, c1), false, bv, (short)0, acc1, false, false);
        }
    }
    const float bi = bias[col];
#pragma unroll
    for (int r = 0; r < 8; ++r) {
        const int mm = r + 8 * h;
        Z[(size_t)(rowBase + mm) * COUT + col]      = acc0[r] + bi;
        Z[(size_t)(rowBase + 16 + mm) * COUT + col] = acc1[r] + bi;
    }
}

// ---------------------------------------------------------------------------
// GEMM2: conv2 (k=3, pad 1). A col (dk*256+ci) = z1[n][t+dk-1][ci], zero-pad t.
// 6 K-panels of 128.
// ---------------------------------------------------------------------------
__global__ void gemm2_kernel(const float* __restrict__ Z1,
                             const unsigned short* __restrict__ WB,
                             const float* __restrict__ bias,
                             float* __restrict__ Z2) {
    __shared__ __align__(16) unsigned short As[32][LDSP];
    const int tid  = threadIdx.x;
    const int lane = tid & 31, wave = tid >> 5;
    const int rowBase = blockIdx.x * 32;
    const int coBase  = blockIdx.y * 128 + wave * 16;
    const int m = lane & 15, h = lane >> 4;
    const int col = coBase + m;

    const int rl = tid >> 3;
    const int kq = (tid & 7) * 16;
    const int grow = rowBase + rl;
    const int n = grow / TSTEP, t = grow - n * TSTEP;

    v8f acc0 = {}, acc1 = {};
    for (int g2 = 0; g2 < 6; ++g2) {
        __syncthreads();
#pragma unroll
        for (int half = 0; half < 2; ++half) {
            const int kglob = g2 * 128 + kq + half * 8;
            const int dk = kglob >> 8, ci = kglob & 255;
            const int ts = t + dk - 1;
            v8f fv = {};
            if ((unsigned)ts < (unsigned)TSTEP)
                fv = *(const v8f*)(Z1 + ((size_t)(n * TSTEP + ts)) * COUT + ci);
            *(v8u16*)&As[rl][kq + half * 8] = cvt8(fv);
        }
        __syncthreads();
#pragma unroll
        for (int s = 0; s < 4; ++s) {
            const int kk = s * 32;
            v8u16 a0 = *(const v8u16*)&As[m][kk + 8 * h];
            v8u16 a1 = *(const v8u16*)&As[m][kk + 16 + 8 * h];
            v8u16 c0 = *(const v8u16*)&As[16 + m][kk + 8 * h];
            v8u16 c1 = *(const v8u16*)&As[16 + m][kk + 16 + 8 * h];
            const unsigned short* bp = WB + (size_t)col * K2TOT + g2 * 128 + kk + 8 * h;
            v16bf bv = packbf(*(const v8u16*)bp, *(const v8u16*)(bp + 16));
            acc0 = __builtin_amdgcn_wmma_f32_16x16x32_bf16(
                false, packbf(a0, a1), false, bv, (short)0, acc0, false, false);
            acc1 = __builtin_amdgcn_wmma_f32_16x16x32_bf16(
                false, packbf(c0, c1), false, bv, (short)0, acc1, false, false);
        }
    }
    const float bi = bias[col];
#pragma unroll
    for (int r = 0; r < 8; ++r) {
        const int mm = r + 8 * h;
        Z2[(size_t)(rowBase + mm) * COUT + col]      = acc0[r] + bi;
        Z2[(size_t)(rowBase + 16 + mm) * COUT + col] = acc1[r] + bi;
    }
}

// ---------------------------------------------------------------------------
// GEMM3: dense w1 on bf16 spikes. A already bf16 -> direct per-lane fragment
// loads from global (weights L2-resident). Fully unrolled, 16 WMMAs/wave.
// ---------------------------------------------------------------------------
__global__ void gemm3_kernel(const unsigned short* __restrict__ S,
                             const unsigned short* __restrict__ WB,
                             float* __restrict__ Z3) {
    const int tid  = threadIdx.x;
    const int lane = tid & 31, wave = tid >> 5;
    const int rowBase = blockIdx.x * 32;
    const int coBase  = blockIdx.y * 128 + wave * 16;
    const int m = lane & 15, h = lane >> 4;
    const int col = coBase + m;
    const unsigned short* arow0 = S + (size_t)(rowBase + m) * COUT;
    const unsigned short* arow1 = S + (size_t)(rowBase + 16 + m) * COUT;

    v8f acc0 = {}, acc1 = {};
#pragma unroll
    for (int c = 0; c < 8; ++c) {
        const int kk = c * 32;
        v8u16 a0 = *(const v8u16*)(arow0 + kk + 8 * h);
        v8u16 a1 = *(const v8u16*)(arow0 + kk + 16 + 8 * h);
        v8u16 c0 = *(const v8u16*)(arow1 + kk + 8 * h);
        v8u16 c1 = *(const v8u16*)(arow1 + kk + 16 + 8 * h);
        const unsigned short* bp = WB + (size_t)col * COUT + kk + 8 * h;
        v16bf bv = packbf(*(const v8u16*)bp, *(const v8u16*)(bp + 16));
        acc0 = __builtin_amdgcn_wmma_f32_16x16x32_bf16(
            false, packbf(a0, a1), false, bv, (short)0, acc0, false, false);
        acc1 = __builtin_amdgcn_wmma_f32_16x16x32_bf16(
            false, packbf(c0, c1), false, bv, (short)0, acc1, false, false);
    }
#pragma unroll
    for (int r = 0; r < 8; ++r) {
        const int mm = r + 8 * h;
        Z3[(size_t)(rowBase + mm) * COUT + col]      = acc0[r];
        Z3[(size_t)(rowBase + 16 + mm) * COUT + col] = acc1[r];
    }
}

// ---------------------------------------------------------------------------
// CUBA LIF scans (state in registers, 63 steps). Spikes stored as bf16 0/1.
// ---------------------------------------------------------------------------
__global__ void cuba_enc_kernel(const float* __restrict__ Z,
                                unsigned short* __restrict__ S) {
    int idx = blockIdx.x * 256 + threadIdx.x;       // 512*256
    int n = idx >> 8, c = idx & 255;
    const float* zp = Z + (size_t)n * TSTEP * COUT + c;
    unsigned short* sp = S + (size_t)n * TSTEP * COUT + c;
    float cur = 0.f, v = 0.f;
    for (int t = 0; t < TSTEP; ++t) {
        float xt = zp[(size_t)t * COUT];
        cur = 0.1f * cur + xt;                      // current_decay = 0.9
        v   = 0.1f * v + cur;                       // voltage_decay = 0.9
        float s = (v >= 0.3f) ? 1.f : 0.f;
        v *= (1.f - s);
        sp[(size_t)t * COUT] = (s != 0.f) ? (unsigned short)0x3F80u : (unsigned short)0u;
    }
}

__global__ void cuba_dense_kernel(const float* __restrict__ Z,
                                  unsigned short* __restrict__ S) {
    int idx = blockIdx.x * 256 + threadIdx.x;
    int n = idx >> 8, c = idx & 255;
    const float* zp = Z + (size_t)n * TSTEP * COUT + c;
    unsigned short* sp = S + (size_t)n * TSTEP * COUT + c;
    float v = 0.f;                                   // current_decay=1 -> i = x
    for (int t = 0; t < TSTEP; ++t) {
        float xt = zp[(size_t)t * COUT];
        v = 0.9f * v + xt;                           // voltage_decay = 0.1
        float s = (v >= 0.1f) ? 1.f : 0.f;
        v *= (1.f - s);
        sp[(size_t)t * COUT] = (s != 0.f) ? (unsigned short)0x3F80u : (unsigned short)0u;
    }
}

// ---------------------------------------------------------------------------
// Head: w2 [2,256] per timestep + LIF. One wave per (n, class).
// ---------------------------------------------------------------------------
__global__ void head_kernel(const unsigned short* __restrict__ S2,
                            const float* __restrict__ w2,
                            float* __restrict__ out) {
    const int tid = threadIdx.x, lane = tid & 31, wave = tid >> 5;
    const int gw = blockIdx.x * 8 + wave;           // 0..1023
    const int n = gw >> 1, o = gw & 1;
    float wc[8];
#pragma unroll
    for (int j = 0; j < 8; ++j) wc[j] = w2[o * COUT + lane * 8 + j];
    float v = 0.f;
    for (int t = 0; t < TSTEP; ++t) {
        const unsigned short* sp = S2 + ((size_t)(n * TSTEP + t)) * COUT + lane * 8;
        v8u16 sv = *(const v8u16*)sp;
        float p = 0.f;
#pragma unroll
        for (int j = 0; j < 8; ++j) p += sv[j] ? wc[j] : 0.f;
#pragma unroll
        for (int off = 16; off; off >>= 1) p += __shfl_xor(p, off, 32);
        v = 0.9f * v + p;
        float s = (v >= 0.1f) ? 1.f : 0.f;
        v *= (1.f - s);
        if (lane == 0) out[((size_t)n * 2 + o) * TSTEP + t] = s;
    }
}

// ---------------------------------------------------------------------------
// Launch
// ---------------------------------------------------------------------------
extern "C" void kernel_launch(void* const* d_in, const int* in_sizes, int n_in,
                              void* d_out, int out_size, void* d_ws, size_t ws_size,
                              hipStream_t stream) {
    (void)in_sizes; (void)n_in; (void)out_size; (void)ws_size;
    const float* x       = (const float*)d_in[0];
    const float* conv1_w = (const float*)d_in[1];
    const float* conv1_b = (const float*)d_in[2];
    const float* bn1_g   = (const float*)d_in[3];
    const float* bn1_b   = (const float*)d_in[4];
    const float* bn1_m   = (const float*)d_in[5];
    const float* bn1_v   = (const float*)d_in[6];
    const float* conv2_w = (const float*)d_in[7];
    const float* conv2_b = (const float*)d_in[8];
    const float* bn2_g   = (const float*)d_in[9];
    const float* bn2_b   = (const float*)d_in[10];
    const float* bn2_m   = (const float*)d_in[11];
    const float* bn2_v   = (const float*)d_in[12];
    const float* w1      = (const float*)d_in[13];
    const float* w2      = (const float*)d_in[14];

    char* w = (char*)d_ws;
    unsigned short* WB1   = (unsigned short*)(w);                 // 1,048,576 B
    float*          bias1 = (float*)(w + 1048576);                // 4 KB slot
    unsigned short* WB2   = (unsigned short*)(w + 1052672);       // 393,216 B
    float*          bias2 = (float*)(w + 1445888);                // 4 KB slot
    unsigned short* WB3   = (unsigned short*)(w + 1449984);       // 131,072 B
    float*          bufA  = (float*)(w + 1581056);                // 33,030,144 B (z1 / z3)
    float*          bufB  = (float*)(w + 34611200);               // 33,030,144 B (z2)
    unsigned short* Sbuf  = (unsigned short*)(w + 67641344);      // 16,515,072 B (s1 / s2)
    // total: 84,156,416 bytes

    prep_conv1<<<2048, 256, 0, stream>>>(conv1_w, conv1_b, bn1_g, bn1_b, bn1_m, bn1_v, WB1, bias1);
    prep_conv2<<<768,  256, 0, stream>>>(conv2_w, conv2_b, bn2_g, bn2_b, bn2_m, bn2_v, WB2, bias2);
    prep_dense<<<256,  256, 0, stream>>>(w1, WB3);

    dim3 g(ROWS / 32, 2);                              // 1008 x 2, block = 8 waves
    gemm1_kernel<<<g, 256, 0, stream>>>(x, WB1, bias1, bufA);
    gemm2_kernel<<<g, 256, 0, stream>>>(bufA, WB2, bias2, bufB);
    cuba_enc_kernel<<<512, 256, 0, stream>>>(bufB, Sbuf);
    gemm3_kernel<<<g, 256, 0, stream>>>(Sbuf, WB3, bufA);
    cuba_dense_kernel<<<512, 256, 0, stream>>>(bufA, Sbuf);
    head_kernel<<<128, 256, 0, stream>>>(Sbuf, w2, (float*)d_out);
}
